// EmbodiedMatcher_81827716923407
// MI455X (gfx1250) — matrix operand order repaired and checked by reading, not compiled, not run on hardware
//
#include <hip/hip_runtime.h>

typedef __attribute__((ext_vector_type(16))) _Float16 v16h;
typedef __attribute__((ext_vector_type(8)))  float    v8f;

#define B_  4
#define N_  256
#define M_  64
#define P_  50000
#define S_  2
#define BIG_ 1e8f

#define LOG2E_ 1.4426950408889634f
#define LN2_   0.6931471805599453f

#define KSPLIT       16
#define KSTEPS_TOTAL ((P_ + 31) / 32)                       /* 1563 */
#define KSTEPS_FULL  (P_ / 32)                              /* 1562 (no OOB) */
#define KSTEPS_PER   ((KSTEPS_TOTAL + KSPLIT - 1) / KSPLIT) /* 98   */

/* workspace layout (floats) */
#define WS_DOT_SIG 0
#define WS_DOT_PM  (B_ * N_ * M_)
#define WS_RS_SIG  (2 * B_ * N_ * M_)
#define WS_RS_NEG  (2 * B_ * N_ * M_ + B_ * N_)
#define WS_CS_M    (2 * B_ * N_ * M_ + 2 * B_ * N_)
#define WS_TOTAL   (2 * B_ * N_ * M_ + 2 * B_ * N_ + B_ * M_)

__global__ void matcher_zero_kernel(float* __restrict__ ws, int count) {
    int i = blockIdx.x * blockDim.x + threadIdx.x;
    if (i < count) ws[i] = 0.0f;
}

__device__ __forceinline__ void ld4(float* dst, const float* base, int k) {
    const float4 t = *(const float4*)(base + k);
    dst[0] = t.x; dst[1] = t.y; dst[2] = t.z; dst[3] = t.w;
}

__device__ __forceinline__ void ld4_guard(float* dst, const float* base,
                                          int k, bool& valid) {
    valid = (k + 4) <= P_;   // 16B-aligned chunks; P_%4==0 -> all-in or all-out
    if (valid) {
        const float4 t = *(const float4*)(base + k);
        dst[0] = t.x; dst[1] = t.y; dst[2] = t.z; dst[3] = t.w;
    } else {
        dst[0] = dst[1] = dst[2] = dst[3] = 0.0f;
    }
}

/* One K-step (32 deep).
   FULL: unconditional loads (1562 of 1563 steps).
   CSUM: accumulate mask column-sums (only the ntile==0 wave needs them). */
template <bool FULL, bool CSUM>
__device__ __forceinline__ void do_step(const float* __restrict__ pmRow,
                                        const float* __restrict__ mBase,
                                        int k0, int g, int r,
                                        v8f (&accS)[4], v8f (&accP)[4],
                                        float& rsS, float& rsN,
                                        float (&csM)[4]) {
    /* ---- A fragments: contiguous k order, k = g*16 + h (B uses the same).
       halves 0..7  = pm[k0 + g*16     .. +7]
       halves 8..15 = pm[k0 + g*16 + 8 .. +7]                                */
    float av[16];
    bool  avd[4] = {true, true, true, true};
    const int ka = k0 + g * 16;
    const int kb = k0 + g * 16 + 8;
    if (FULL) {
        ld4(av + 0,  pmRow, ka);
        ld4(av + 4,  pmRow, ka + 4);
        ld4(av + 8,  pmRow, kb);
        ld4(av + 12, pmRow, kb + 4);
    } else {
        ld4_guard(av + 0,  pmRow, ka,     avd[0]);
        ld4_guard(av + 4,  pmRow, ka + 4, avd[1]);
        ld4_guard(av + 8,  pmRow, kb,     avd[2]);
        ld4_guard(av + 12, pmRow, kb + 4, avd[3]);
    }
    v16h aS, aP;
#pragma unroll
    for (int h = 0; h < 16; ++h) {
        const float x   = av[h];
        /* raw HW transcendentals: arg of exp2 <= 0 (result in (0,1]),
           arg of log2 in [1,2] -> no denorm/range fixups required        */
        const float t   = __builtin_amdgcn_exp2f(-fabsf(x) * LOG2E_); // e^{-|x|}
        const float inv = __builtin_amdgcn_rcpf(1.0f + t);            // v_rcp_f32
        const float sg  = (x >= 0.0f) ? inv : t * inv;                // sigmoid
        const float sp  = fmaxf(x, 0.0f)
                        + __builtin_amdgcn_logf(1.0f + t) * LN2_;     // softplus
        if (FULL || avd[h >> 2]) { rsS += sg; rsN += sp; }
        aS[h] = (_Float16)sg;
        aP[h] = (_Float16)x;
    }
    if (FULL) __builtin_prefetch(pmRow + k0 + 512, 0, 1);  // global_prefetch_b8

    /* ---- B fragments + WMMA: same contiguous k order as A, lane (g, n=r):
       halves 0..15 = m[n][k0 + g*16 .. +15]                                 */
#pragma unroll
    for (int mt = 0; mt < 4; ++mt) {
        const float* mRow = mBase + (size_t)(mt * 16 + r) * P_;
        const int kb0 = k0 + g * 16;
        float bv[16];
        if (FULL) {
            ld4(bv + 0,  mRow, kb0);
            ld4(bv + 4,  mRow, kb0 + 4);
            ld4(bv + 8,  mRow, kb0 + 8);
            ld4(bv + 12, mRow, kb0 + 12);
        } else {
            bool dummy;
            ld4_guard(bv + 0,  mRow, kb0,      dummy);
            ld4_guard(bv + 4,  mRow, kb0 + 4,  dummy);
            ld4_guard(bv + 8,  mRow, kb0 + 8,  dummy);
            ld4_guard(bv + 12, mRow, kb0 + 12, dummy);
        }
        v16h bM;
        float cs = 0.0f;
#pragma unroll
        for (int h = 0; h < 16; ++h) {
            bM[h] = (_Float16)bv[h];   // binary mask: exact in f16; OOB tail is 0
            if (CSUM) cs += bv[h];
        }
        if (CSUM) csM[mt] += cs;
        accS[mt] = __builtin_amdgcn_wmma_f32_16x16x32_f16(
            false, aS, false, bM, (short)0, accS[mt], false, false);
        accP[mt] = __builtin_amdgcn_wmma_f32_16x16x32_f16(
            false, aP, false, bM, (short)0, accP[mt], false, false);
    }
}

__global__ __launch_bounds__(256)
void matcher_gemm_kernel(const float* __restrict__ pm,   // [B,N,P]
                         const float* __restrict__ m,    // [B,M,P]
                         float* __restrict__ ws) {
    const int lane  = threadIdx.x & 31;
    const int wave  = threadIdx.x >> 5;
    const int g     = lane >> 4;      // lane group (0/1)
    const int r     = lane & 15;      // row/col within tile
    const int ntile = blockIdx.y * 8 + wave;   // 0..15
    const int batch = blockIdx.z;
    const int ks    = blockIdx.x;

    const float* pmRow = pm + (size_t)batch * N_ * P_ + (size_t)(ntile * 16 + r) * P_;
    const float* mBase = m  + (size_t)batch * M_ * P_;

    v8f accS[4] = {{}, {}, {}, {}};
    v8f accP[4] = {{}, {}, {}, {}};
    float rsS = 0.0f, rsN = 0.0f;
    float csM[4] = {0.0f, 0.0f, 0.0f, 0.0f};

    const int sBeg     = ks * KSTEPS_PER;
    const int sEnd     = min(sBeg + KSTEPS_PER, KSTEPS_TOTAL);
    const int sFullEnd = min(sEnd, KSTEPS_FULL);
    const int sTail    = max(sBeg, sFullEnd);

    if (ntile == 0) {    // wave-uniform: this wave also reduces mask column-sums
        for (int s = sBeg; s < sFullEnd; ++s)
            do_step<true, true>(pmRow, mBase, s * 32, g, r, accS, accP, rsS, rsN, csM);
        for (int s = sTail; s < sEnd; ++s)
            do_step<false, true>(pmRow, mBase, s * 32, g, r, accS, accP, rsS, rsN, csM);
    } else {             // common path: no column-sum VALU work in the hot loop
        for (int s = sBeg; s < sFullEnd; ++s)
            do_step<true, false>(pmRow, mBase, s * 32, g, r, accS, accP, rsS, rsN, csM);
        for (int s = sTail; s < sEnd; ++s)
            do_step<false, false>(pmRow, mBase, s * 32, g, r, accS, accP, rsS, rsN, csM);
    }

    /* ---- epilogue: partial-sum reduction into workspace ---- */
    float* dotS = ws + WS_DOT_SIG + (size_t)batch * N_ * M_;
    float* dotP = ws + WS_DOT_PM  + (size_t)batch * N_ * M_;
#pragma unroll
    for (int mt = 0; mt < 4; ++mt) {
#pragma unroll
        for (int j = 0; j < 8; ++j) {
            // C/D layout: VGPR j, lanes 0-15 -> row j, lanes 16-31 -> row j+8; col = lane%16
            const int row = ntile * 16 + j + 8 * g;
            const int col = mt * 16 + r;
            atomicAdd(&dotS[row * M_ + col], accS[mt][j]);
            atomicAdd(&dotP[row * M_ + col], accP[mt][j]);
        }
    }
    rsS += __shfl_xor(rsS, 16, 32);
    rsN += __shfl_xor(rsN, 16, 32);
    if (g == 0) {
        atomicAdd(&ws[WS_RS_SIG + batch * N_ + ntile * 16 + r], rsS);
        atomicAdd(&ws[WS_RS_NEG + batch * N_ + ntile * 16 + r], rsN);
    }
    if (ntile == 0) {
#pragma unroll
        for (int mt = 0; mt < 4; ++mt) {
            const float c = csM[mt] + __shfl_xor(csM[mt], 16, 32);
            if (g == 0) atomicAdd(&ws[WS_CS_M + batch * M_ + mt * 16 + r], c);
        }
    }
}

__global__ __launch_bounds__(256)
void matcher_finalize_kernel(const float* __restrict__ ps,   // [B,N,S]
                             const int*   __restrict__ sc,   // [B,M]
                             const int*   __restrict__ qgm,  // [B,N,M]
                             const float* __restrict__ ws,
                             float* __restrict__ out) {
    __shared__ float s1[256];
    __shared__ float s2[256];
    const int b  = blockIdx.x / M_;
    const int mc = blockIdx.x % M_;
    const int n  = threadIdx.x;          // one query per thread, N_ == 256

    const float ds = ws[WS_DOT_SIG + (b * N_ + n) * M_ + mc];
    const float dp = ws[WS_DOT_PM  + (b * N_ + n) * M_ + mc];
    const float rs = ws[WS_RS_SIG + b * N_ + n];
    const float rn = ws[WS_RS_NEG + b * N_ + n];
    const float cm = ws[WS_CS_M + b * M_ + mc];

    const float dice  = 1.0f
        - (2.0f * ds + 1.0f) * __builtin_amdgcn_rcpf(rs + cm + 1.0f);
    const float maskc = (rn - dp) * (1.0f / (float)P_); // pos@m+neg@(1-m)=rowsum(neg)-pm@m

    const float sA = ps[(b * N_ + n) * S_ + 0];
    const float sB = ps[(b * N_ + n) * S_ + 1];
    const int  cls = sc[b * M_ + mc];
    const float mx = fmaxf(sA, sB);
    const float eA = __builtin_amdgcn_exp2f((sA - mx) * LOG2E_);
    const float eB = __builtin_amdgcn_exp2f((sB - mx) * LOG2E_);
    const float prob = (cls == 0 ? eA : eB) * __builtin_amdgcn_rcpf(eA + eB);

    float c = -prob + maskc + dice;
    c = qgm[(b * N_ + n) * M_ + mc] ? c : BIG_;

    /* two smallest over the column (k = 2 since N > 1) */
    s1[n] = c;
    s2[n] = 3.4e38f;
    __syncthreads();
    for (int off = 128; off > 0; off >>= 1) {
        if (n < off) {
            const float a1 = s1[n],       a2 = s2[n];
            const float b1 = s1[n + off], b2 = s2[n + off];
            s1[n] = fminf(a1, b1);
            s2[n] = fminf(fmaxf(a1, b1), fminf(a2, b2));
        }
        __syncthreads();
    }
    const float kth = s2[0];

    out[(b * N_ + n) * M_ + mc] = c;
    out[B_ * N_ * M_ + (b * N_ + n) * M_ + mc] = (c < kth) ? 1.0f : 0.0f;
}

extern "C" void kernel_launch(void* const* d_in, const int* in_sizes, int n_in,
                              void* d_out, int out_size, void* d_ws, size_t ws_size,
                              hipStream_t stream) {
    const float* pm  = (const float*)d_in[0];   // pred_masks  [B,N,P]
    const float* psc = (const float*)d_in[1];   // pred_scores [B,N,S]
    const float* msk = (const float*)d_in[2];   // masks       [B,M,P]
    const int*   sc  = (const int*)d_in[3];     // scores      [B,M]
    const int*   qgm = (const int*)d_in[4];     // query_gt_mask [B,N,M]
    float* ws  = (float*)d_ws;
    float* out = (float*)d_out;

    matcher_zero_kernel<<<(WS_TOTAL + 255) / 256, 256, 0, stream>>>(ws, WS_TOTAL);

    dim3 grid(KSPLIT, N_ / (16 * 8), B_);   // (16, 2, 4) -> 128 blocks x 8 waves
    matcher_gemm_kernel<<<grid, 256, 0, stream>>>(pm, msk, ws);

    matcher_finalize_kernel<<<B_ * M_, 256, 0, stream>>>(psc, sc, qgm, ws, out);
}
